// TTNModel_36747740185173
// MI455X (gfx1250) — matrix operand order repaired and checked by reading, not compiled, not run on hardware
//
#include <hip/hip_runtime.h>
#include <hip/hip_bf16.h>
#include <stddef.h>

// ---------------------------------------------------------------------------
// TTN forward on gfx1250 (MI455X): fused tree contraction, bf16 WMMA core.
//   h kept in LDS (bf16, ping-pong). Weights pre-swizzled into WMMA-fragment
//   order in d_ws so each lane loads its 16-element B fragment as one
//   contiguous 32B read (2x global_load_b128) per v_wmma_f32_16x16x32_bf16.
// ---------------------------------------------------------------------------

typedef __attribute__((ext_vector_type(16))) __bf16 bf16x16;
typedef __attribute__((ext_vector_type(8)))  float  f32x8;

#define TTN_BATCH   16384
#define ROWS_PER_WG 32
#define THREADS     256
#define NWAVES      8

// Packed bf16 weight offsets (elements) inside d_ws.
// Packed layer size = nodes * ntiles * (K/32) * 32lanes * 16elems
#define OFF_W0 0u         // 1  node, ntiles=1, K=1024 ->  16384
#define OFF_W1 16384u     // 2  nodes, ntiles=2, K=1024 ->  65536
#define OFF_W2 81920u     // 4  -> 131072
#define OFF_W3 212992u    // 8  -> 262144
#define OFF_W4 475136u    // 16 -> 524288
#define OFF_W5 999424u    // 32 nodes, ntiles=2, K=256 -> 262144
// total: 1,261,568 bf16 elements (~2.4 MB of d_ws)

// Swizzle f32 weights [node][K][O] into per-lane WMMA B fragments:
//   dst[((node*ntiles+nt)*nslices+s)*512 + lane*16 + e]
//     = W[node][s*32 + (e<8 ? e : e+8) + (lane>=16 ? 8:0)][nt*16 + (lane&15)]   (0 if col>=O)
__global__ void pack_weights_bf16(const float* __restrict__ src,
                                  __bf16* __restrict__ dst,
                                  int nodes, int K, int O, int ntiles, int total) {
  int t = blockIdx.x * blockDim.x + threadIdx.x;
  const int stride = gridDim.x * blockDim.x;
  const int nslices = K >> 5;
  for (; t < total; t += stride) {
    const int idx  = t & 511;
    const int frag = t >> 9;
    const int lane = idx >> 4;
    const int e    = idx & 15;
    const int s    = frag % nslices;
    const int f2   = frag / nslices;
    const int nt   = f2 % ntiles;
    const int node = f2 / ntiles;
    const int kk = (s << 5) + ((e < 8) ? e : (e + 8)) + ((lane >= 16) ? 8 : 0);
    const int o  = nt * 16 + (lane & 15);
    const float v = (o < O) ? src[((size_t)node * K + kk) * O + o] : 0.0f;
    dst[t] = (__bf16)v;
  }
}

// One 16x16 output tile per wave; A = on-the-fly outer product (left x right)
// built from LDS bf16, B = one 32B fragment load per slice from packed weights.
__device__ __forceinline__ void wmma_layer(const __bf16* __restrict__ wpacked,
                                           const __bf16* __restrict__ src,
                                           __bf16* __restrict__ dst,
                                           const int nodes, const int dinShift,
                                           const int O,
                                           const int lane, const int wave)
{
  const int din     = 1 << dinShift;
  const int K       = din * din;
  const int nslices = K >> 5;          // K-slices of 32
  const int ntiles  = O >> 4;          // 16-wide N tiles
  const int tasks   = nodes * 2 * ntiles;
  const int laneHi  = lane >> 4;       // 0 / 1
  const int col     = lane & 15;
  const bf16x16* __restrict__ wfrag = (const bf16x16*)wpacked;

  for (int task = wave; task < tasks; task += NWAVES) {
    const int node  = task / (2 * ntiles);
    const int rem   = task - node * (2 * ntiles);
    const int mtile = rem / ntiles;
    const int nt    = rem - mtile * ntiles;

    const int brow = mtile * 16 + (lane & 15);
    const __bf16* lp = src + brow * 1024 + ((2 * node) << dinShift);
    const __bf16* rp = lp + din;
    const size_t fbase = ((size_t)(node * ntiles + nt) * nslices) * 32 + lane;
    __builtin_prefetch(&wfrag[fbase], 0, 1);   // global_prefetch_b8

    f32x8 c = {0.f, 0.f, 0.f, 0.f, 0.f, 0.f, 0.f, 0.f};
#pragma unroll 2
    for (int s = 0; s < nslices; ++s) {
      const int k0 = s << 5;
      bf16x16 a;
#pragma unroll
      for (int e = 0; e < 16; ++e) {
        // A 16x32 bf16 layout: lanes 0-15 hold K {0..7,16..23}, lanes 16-31 +8
        const int klA = ((e < 8) ? e : (e + 8)) + (laneHi << 3);
        const int kk  = k0 + klA;
        a[e] = (__bf16)((float)lp[kk >> dinShift] * (float)rp[kk & (din - 1)]);
      }
      const bf16x16 bmat = wfrag[fbase + (size_t)s * 32];
      c = __builtin_amdgcn_wmma_f32_16x16x32_bf16(false, a, false, bmat,
                                                  (short)0, c, false, false);
    }
    // D (f32 16x16): VGPR d -> M = d (lanes 0-15) / d+8 (lanes 16-31), N = col
    __bf16* dp = dst + node * O + nt * 16 + col;
#pragma unroll
    for (int d = 0; d < 8; ++d) {
      const int row = mtile * 16 + (laneHi << 3) + d;
      dp[row * 1024] = (__bf16)c[d];
    }
  }
}

// Root layer: 1 node, K = 1024, O = 10 (zero-padded to 16 at pack time).
// K split 4-ways across waves; partials reduced deterministically in LDS.
__device__ __forceinline__ void wmma_root(const __bf16* __restrict__ w0packed,
                                          const __bf16* __restrict__ src,
                                          float* __restrict__ cacc,
                                          const int lane, const int wave)
{
  const int laneHi = lane >> 4;
  const int col    = lane & 15;
  const int mtile  = wave & 1;
  const int kq     = wave >> 1;              // K-quarter 0..3
  const int brow   = mtile * 16 + (lane & 15);
  const __bf16* lp = src + brow * 1024;      // node 0 (left), entries 0..31
  const __bf16* rp = lp + 32;                // node 1 (right), entries 32..63
  const bf16x16* __restrict__ wfrag = (const bf16x16*)w0packed;

  f32x8 c = {0.f, 0.f, 0.f, 0.f, 0.f, 0.f, 0.f, 0.f};
#pragma unroll 2
  for (int s = kq * 8; s < kq * 8 + 8; ++s) {
    const int k0 = s << 5;
    bf16x16 a;
#pragma unroll
    for (int e = 0; e < 16; ++e) {
      const int klA = ((e < 8) ? e : (e + 8)) + (laneHi << 3);
      const int kk  = k0 + klA;
      a[e] = (__bf16)((float)lp[kk >> 5] * (float)rp[kk & 31]);
    }
    const bf16x16 bmat = wfrag[(size_t)s * 32 + lane];
    c = __builtin_amdgcn_wmma_f32_16x16x32_bf16(false, a, false, bmat,
                                                (short)0, c, false, false);
  }
#pragma unroll
  for (int d = 0; d < 8; ++d) {
    const int row = mtile * 16 + (laneHi << 3) + d;
    cacc[(kq * ROWS_PER_WG + row) * 16 + col] = c[d];   // unique slot: no atomics
  }
}

__launch_bounds__(THREADS, 1)
__global__ void ttn_fused_kernel(const float* __restrict__ x,
                                 const float* __restrict__ w6f,
                                 const float* __restrict__ w7f,
                                 const __bf16* __restrict__ wbf,
                                 float* __restrict__ out)
{
  __shared__ __bf16 bufA[ROWS_PER_WG * 1024];       // 64 KB
  __shared__ __bf16 bufB[ROWS_PER_WG * 1024];       // 64 KB
  __shared__ float  cacc[4 * ROWS_PER_WG * 16];     //  8 KB

  const int tid  = threadIdx.x;
  const int lane = tid & 31;
  const int wave = tid >> 5;
  const int b0   = blockIdx.x * ROWS_PER_WG;

  // Stage in x tile: (b, 256 features, 2 phys) -> bf16, row stride 1024
  for (int i = tid; i < ROWS_PER_WG * 512; i += THREADS) {
    const int row = i >> 9;
    const int idx = i & 511;
    bufA[row * 1024 + idx] = (__bf16)x[(size_t)(b0 + row) * 512 + idx];
  }
  __syncthreads();

  // Layer 7 (128 nodes, K=4, N=4) -- VALU f32
  for (int p = tid; p < ROWS_PER_WG * 128; p += THREADS) {
    const int row = p >> 7;
    const int n   = p & 127;
    const __bf16* hr = bufA + row * 1024 + 4 * n;
    const float l0 = (float)hr[0], l1 = (float)hr[1];
    const float r0 = (float)hr[2], r1 = (float)hr[3];
    const float* wn = w7f + n * 16;                  // w7[n][l][r][o]
    __bf16* orow = bufB + row * 1024 + n * 4;
#pragma unroll
    for (int o = 0; o < 4; ++o) {
      orow[o] = (__bf16)(l0 * r0 * wn[o]     + l0 * r1 * wn[4 + o] +
                         l1 * r0 * wn[8 + o] + l1 * r1 * wn[12 + o]);
    }
  }
  __syncthreads();

  // Layer 6 (64 nodes, K=16, N=16) -- VALU f32
  for (int p = tid; p < ROWS_PER_WG * 64; p += THREADS) {
    const int row = p >> 6;
    const int n   = p & 63;
    const __bf16* hr = bufB + row * 1024 + 8 * n;
    float L[4], R[4];
#pragma unroll
    for (int i = 0; i < 4; ++i) { L[i] = (float)hr[i]; R[i] = (float)hr[4 + i]; }
    const float* wn = w6f + n * 256;                 // w6[n][l][r][o]
    __bf16* orow = bufA + row * 1024 + n * 16;
#pragma unroll
    for (int o = 0; o < 16; ++o) {
      float acc = 0.f;
#pragma unroll
      for (int l = 0; l < 4; ++l)
#pragma unroll
        for (int r = 0; r < 4; ++r)
          acc += L[l] * R[r] * wn[l * 64 + r * 16 + o];
      orow[o] = (__bf16)acc;
    }
  }
  __syncthreads();

  // WMMA layers: (nodes, log2 din, O)
  wmma_layer(wbf + OFF_W5, bufA, bufB, 32, 4, 32, lane, wave); __syncthreads();
  wmma_layer(wbf + OFF_W4, bufB, bufA, 16, 5, 32, lane, wave); __syncthreads();
  wmma_layer(wbf + OFF_W3, bufA, bufB,  8, 5, 32, lane, wave); __syncthreads();
  wmma_layer(wbf + OFF_W2, bufB, bufA,  4, 5, 32, lane, wave); __syncthreads();
  wmma_layer(wbf + OFF_W1, bufA, bufB,  2, 5, 32, lane, wave); __syncthreads();

  // Root layer (1 node, K=1024, O=10 padded to 16)
  wmma_root(wbf + OFF_W0, bufB, cacc, lane, wave);
  __syncthreads();

  // Deterministic 4-way K-partial reduction + f32 output store
  for (int i = tid; i < ROWS_PER_WG * 10; i += THREADS) {
    const int row = i / 10;
    const int o   = i - row * 10;
    const float acc = cacc[(0 * ROWS_PER_WG + row) * 16 + o]
                    + cacc[(1 * ROWS_PER_WG + row) * 16 + o]
                    + cacc[(2 * ROWS_PER_WG + row) * 16 + o]
                    + cacc[(3 * ROWS_PER_WG + row) * 16 + o];
    out[(size_t)(b0 + row) * 10 + o] = acc;
  }
}

extern "C" void kernel_launch(void* const* d_in, const int* in_sizes, int n_in,
                              void* d_out, int out_size, void* d_ws, size_t ws_size,
                              hipStream_t stream) {
  const float* x = (const float*)d_in[0];
  const float* w[8];
  for (int i = 0; i < 8; ++i) w[i] = (const float*)d_in[1 + i];
  __bf16* wbf = (__bf16*)d_ws;   // needs ~2.5 MB of workspace

  // Pre-swizzle w0..w5 into WMMA-fragment order (bf16) in d_ws.
  //            {src, nodes,  K,  O, ntiles, total packed elems, dst offset}
  const int      pnodes[6] = {1, 2, 4, 8, 16, 32};
  const int      pK[6]     = {1024, 1024, 1024, 1024, 1024, 256};
  const int      pO[6]     = {10, 32, 32, 32, 32, 32};
  const int      pnt[6]    = {1, 2, 2, 2, 2, 2};
  const int      ptot[6]   = {16384, 65536, 131072, 262144, 524288, 262144};
  const unsigned poff[6]   = {OFF_W0, OFF_W1, OFF_W2, OFF_W3, OFF_W4, OFF_W5};
  for (int i = 0; i < 6; ++i) {
    const int blocks = (ptot[i] + THREADS - 1) / THREADS;
    pack_weights_bf16<<<dim3(blocks), dim3(THREADS), 0, stream>>>(
        w[i], wbf + poff[i], pnodes[i], pK[i], pO[i], pnt[i], ptot[i]);
  }

  ttn_fused_kernel<<<dim3(TTN_BATCH / ROWS_PER_WG), dim3(THREADS), 0, stream>>>(
      x, w[6], w[7], wbf, (float*)d_out);

  (void)in_sizes; (void)n_in; (void)out_size; (void)ws_size;
}
